// ScaledDotProductAttention_13924283973657
// MI455X (gfx1250) — matrix operand order repaired and checked by reading, not compiled, not run on hardware
//
#include <hip/hip_runtime.h>
#include <math.h>

// ---------------------------------------------------------------- types
typedef __attribute__((ext_vector_type(16))) __bf16 v16bf;
typedef __attribute__((ext_vector_type(8)))  __bf16 v8bf;
typedef __attribute__((ext_vector_type(4)))  __bf16 v4bf;
typedef __attribute__((ext_vector_type(8)))  float  v8f;
typedef __attribute__((ext_vector_type(4)))  float  v4f;
typedef __attribute__((ext_vector_type(4)))  unsigned int u32x4;
typedef __attribute__((ext_vector_type(8)))  int i32x8;
typedef __attribute__((ext_vector_type(4)))  int i32x4;

#define D_MODEL 1024
#define SEQ     2048
#define BATCH   4
#define HEADS   16
#define DHEAD   64

__device__ __forceinline__ __bf16 f32_to_bf16(float f) {
  unsigned u = __builtin_bit_cast(unsigned, f);
  u += 0x7fffu + ((u >> 16) & 1u);                 // round-to-nearest-even
  unsigned short h = (unsigned short)(u >> 16);
  return __builtin_bit_cast(__bf16, h);
}

__device__ __forceinline__ v16bf cat8(v8bf lo, v8bf hi) {
  return __builtin_shufflevector(lo, hi, 0,1,2,3,4,5,6,7,8,9,10,11,12,13,14,15);
}

__device__ __forceinline__ v8f wmma_bf16(v16bf a, v16bf b, v8f c) {
  return __builtin_amdgcn_wmma_f32_16x16x32_bf16(false, a, false, b, (short)0, c,
                                                 false, false);
}

// ---- Tensor Data Mover: 2D bf16 tile load, LDS padding 16B per 64B row ----
// D# per CDNA5 ISA ch.8: group0 {count, lds_addr, global_addr, type=2},
// group1 {data_size=2B, pad_enable, pad_interval=16dw, pad_amount=4dw,
//         tensor dims, tile dims, dim0 stride}.  2D -> groups 2/3 zero.
// This toolchain's builtin is the 6-arg form:
//   (u32x4 g0, i32x8 g1, i32x4 g2, i32x4 g3, i32x8 zero, i32 cpol)
__device__ __forceinline__ void tdm_load_2d(unsigned lds_off,
                                            const __bf16* gptr,
                                            unsigned tensor_d0,
                                            unsigned tensor_d1,
                                            unsigned tile_d0,
                                            unsigned tile_d1,
                                            unsigned d0_stride) {
  unsigned long long ga = (unsigned long long)(size_t)gptr;
  u32x4 g0;
  g0[0] = 1u;                                       // count=1 (valid user D#)
  g0[1] = lds_off;                                  // lds_addr (bytes)
  g0[2] = (unsigned)(ga & 0xffffffffu);             // global_addr[31:0]
  g0[3] = (unsigned)((ga >> 32) & 0x01ffffffu)      // global_addr[56:32]
        | (2u << 30);                               // type = 2 ("image")
  i32x8 g1;
  g1[0] = (int)((1u << 16)                          // data_size = 2 bytes
              | (1u << 20)                          // pad_enable
              | (3u << 22)                          // pad_interval: 16 DWORDs
              | (3u << 25));                        // pad_amount:    4 DWORDs
  g1[1] = (int)((tensor_d0 & 0xffffu) << 16);                 // dim0[15:0]
  g1[2] = (int)(((tensor_d0 >> 16) & 0xffffu)
              | ((tensor_d1 & 0xffffu) << 16));               // dim0[31:16]|dim1[15:0]
  g1[3] = (int)(((tensor_d1 >> 16) & 0xffffu)
              | ((tile_d0 & 0xffffu) << 16));                 // dim1[31:16]|tile0
  g1[4] = (int)(tile_d1 & 0xffffu);                           // tile1 (tile2=0)
  g1[5] = (int)d0_stride;                                     // dim0_stride[31:0]
  g1[6] = 0;
  g1[7] = 0;
  i32x4 gz4 = {0, 0, 0, 0};
  i32x8 gz8 = {0, 0, 0, 0, 0, 0, 0, 0};
  __builtin_amdgcn_tensor_load_to_lds(g0, g1, gz4, gz4, gz8, 0);
}

// ---------------------------------------------------------------- fp32 -> bf16
__global__ void cvt_f32_to_bf16(const float* __restrict__ in,
                                __bf16* __restrict__ out, int n) {
  int i = (blockIdx.x * blockDim.x + threadIdx.x) * 4;
  if (i < n) {
    v4f x = *(const v4f*)(in + i);
    v4bf y;
    y[0] = f32_to_bf16(x[0]); y[1] = f32_to_bf16(x[1]);
    y[2] = f32_to_bf16(x[2]); y[3] = f32_to_bf16(x[3]);
    *(v4bf*)(out + i) = y;
  }
}

// ---------------------------------------------------------------- GEMM core
// Y[m,n] = sum_k A[m,k] * W[n,k] (+bias).  Block 128x128, BK=32.
// 8 waves as 4(m) x 2(n); each wave 32x64 = 2x4 WMMA tiles (8 wmma / K-step).
// A/B tiles staged by TDM into double-buffered padded LDS.
#define BM  128
#define BN  128
#define BK  32
#define LDP 40   // padded LDS row stride in bf16 (64B data + 16B TDM pad)

struct Frag { v16bf af[2]; v16bf bfr[4]; };

__device__ __forceinline__ void gemm_body(const __bf16* Aptr,   // block A base
                                          const __bf16* Bptr,   // block B base
                                          unsigned mrows, unsigned nrows,
                                          __bf16* AsBase, __bf16* BsBase,
                                          int wid, int wm, int wn, int hf, int lr,
                                          v8f acc[2][4]) {
  const bool issuer = (wid == 0);
  const unsigned asOff = (unsigned)(size_t)AsBase;   // LDS aperture: low 32b
  const unsigned bsOff = (unsigned)(size_t)BsBase;
  const unsigned bufA = BM * LDP * 2;                // byte size of one A buf
  const unsigned bufB = BN * LDP * 2;

  if (issuer) {
    tdm_load_2d(asOff, Aptr, D_MODEL, mrows, BK, BM, D_MODEL);
    tdm_load_2d(bsOff, Bptr, D_MODEL, nrows, BK, BN, D_MODEL);
  }
  int buf = 0;
  for (int kk = 0; kk < D_MODEL; kk += BK) {
    const int nkk = kk + BK;
    if (issuer) {
      if (nkk < D_MODEL) {
        tdm_load_2d(asOff + (buf ^ 1) * bufA, Aptr + nkk, D_MODEL, mrows,
                    BK, BM, D_MODEL);
        tdm_load_2d(bsOff + (buf ^ 1) * bufB, Bptr + nkk, D_MODEL, nrows,
                    BK, BN, D_MODEL);
        __builtin_amdgcn_s_wait_tensorcnt(2);   // current pair complete
      } else {
        __builtin_amdgcn_s_wait_tensorcnt(0);
      }
    }
    __syncthreads();

    const __bf16* As = AsBase + buf * (BM * LDP);
    const __bf16* Bs = BsBase + buf * (BN * LDP);
    Frag f;
#pragma unroll
    for (int t = 0; t < 2; ++t) {
      const __bf16* ar = &As[(wm * 32 + t * 16 + lr) * LDP];
      f.af[t] = cat8(*(const v8bf*)(ar + hf * 8),
                     *(const v8bf*)(ar + 16 + hf * 8));
    }
#pragma unroll
    for (int t = 0; t < 4; ++t) {
      const __bf16* br = &Bs[(wn * 64 + t * 16 + lr) * LDP + hf * 8];
      f.bfr[t] = cat8(*(const v8bf*)br, *(const v8bf*)(br + 16));
    }
#pragma unroll
    for (int tm = 0; tm < 2; ++tm)
#pragma unroll
      for (int tn = 0; tn < 4; ++tn)
        acc[tm][tn] = wmma_bf16(f.af[tm], f.bfr[tn], acc[tm][tn]);
    __syncthreads();
    buf ^= 1;
  }
}

// ---------------------------------------------------------------- QKV GEMM
__global__ __launch_bounds__(256)
void gemm_qkv(const __bf16* __restrict__ Xb,    // [8192][1024]
              const __bf16* __restrict__ Wqkv,  // [3072][1024]
              const float* __restrict__ bq, const float* __restrict__ bk,
              const float* __restrict__ bv,
              __bf16* __restrict__ Qb,          // [B][H][S][64]
              __bf16* __restrict__ Kb,          // [B][H][S][64]
              __bf16* __restrict__ Vt) {        // [B][H][64][S]
  __shared__ __align__(128) __bf16 As[2][BM * LDP];
  __shared__ __align__(128) __bf16 Bs[2][BN * LDP];

  const int tid = threadIdx.x, lane = tid & 31, wid = tid >> 5;
  const int wm = wid >> 1, wn = wid & 1;
  const int hf = lane >> 4, lr = lane & 15;
  const int m_blk = blockIdx.y * BM;
  const int n_blk = blockIdx.x * BN;

  v8f acc[2][4] = {};
  gemm_body(Xb + (size_t)m_blk * D_MODEL, Wqkv + (size_t)n_blk * D_MODEL,
            BATCH * SEQ - m_blk, 3 * D_MODEL - n_blk,
            &As[0][0], &Bs[0][0], wid, wm, wn, hf, lr, acc);

  // epilogue: +bias, scatter into head layouts (which is uniform per block)
#pragma unroll
  for (int tm = 0; tm < 2; ++tm) {
#pragma unroll
    for (int tn = 0; tn < 4; ++tn) {
      const int n_g   = n_blk + wn * 64 + tn * 16 + lr;
      const int which = n_g >> 10;            // 0=q 1=k 2=v
      const int nn    = n_g & (D_MODEL - 1);
      const float bias = (which == 0) ? bq[nn] : (which == 1) ? bk[nn] : bv[nn];
      const int h = nn >> 6, d = nn & 63;
#pragma unroll
      for (int r = 0; r < 8; ++r) {
        const int m_g = m_blk + wm * 32 + tm * 16 + r + hf * 8;
        const int b = m_g >> 11, s = m_g & (SEQ - 1);
        const __bf16 o = f32_to_bf16(acc[tm][tn][r] + bias);
        const size_t bh = (size_t)(b * HEADS + h);
        if (which == 0)      Qb[(bh * SEQ + s) * DHEAD + d] = o;
        else if (which == 1) Kb[(bh * SEQ + s) * DHEAD + d] = o;
        else                 Vt[(bh * DHEAD + d) * SEQ + s] = o;
      }
    }
  }
}

// ---------------------------------------------------------------- O-proj GEMM
__global__ __launch_bounds__(256)
void gemm_out(const __bf16* __restrict__ AO,   // [8192][1024]
              const __bf16* __restrict__ Wo,   // [1024][1024]
              const float* __restrict__ bo,
              float* __restrict__ out) {       // [8192][1024] f32
  __shared__ __align__(128) __bf16 As[2][BM * LDP];
  __shared__ __align__(128) __bf16 Bs[2][BN * LDP];

  const int tid = threadIdx.x, lane = tid & 31, wid = tid >> 5;
  const int wm = wid >> 1, wn = wid & 1;
  const int hf = lane >> 4, lr = lane & 15;
  const int m_blk = blockIdx.y * BM;
  const int n_blk = blockIdx.x * BN;

  v8f acc[2][4] = {};
  gemm_body(AO + (size_t)m_blk * D_MODEL, Wo + (size_t)n_blk * D_MODEL,
            BATCH * SEQ - m_blk, D_MODEL - n_blk,
            &As[0][0], &Bs[0][0], wid, wm, wn, hf, lr, acc);

#pragma unroll
  for (int tm = 0; tm < 2; ++tm) {
#pragma unroll
    for (int tn = 0; tn < 4; ++tn) {
      const int n_g = n_blk + wn * 64 + tn * 16 + lr;
      const float bias = bo[n_g];
#pragma unroll
      for (int r = 0; r < 8; ++r) {
        const int m_g = m_blk + wm * 32 + tm * 16 + r + hf * 8;
        out[(size_t)m_g * D_MODEL + n_g] = acc[tm][tn][r] + bias;
      }
    }
  }
}

// ---------------------------------------------------------------- attention
// Flash-style: each wave owns 16 query rows, streams K/V in 32-wide chunks.
__global__ __launch_bounds__(256)
void attention(const __bf16* __restrict__ Qb, const __bf16* __restrict__ Kb,
               const __bf16* __restrict__ Vt, __bf16* __restrict__ AO) {
  __shared__ __align__(16) __bf16 Plds[8][16 * LDP];

  const int tid  = threadIdx.x;
  const int lane = tid & 31, wid = tid >> 5;
  const int hf = lane >> 4, lr = lane & 15;
  const int bh = blockIdx.y;                       // b*HEADS + h
  const int q0 = blockIdx.x * 128 + wid * 16;
  const size_t qkBase = (size_t)bh * SEQ * DHEAD;
  const size_t vtBase = (size_t)bh * DHEAD * SEQ;

  v16bf aq[2];
  {
    const __bf16* qr = Qb + qkBase + (size_t)(q0 + lr) * DHEAD;
#pragma unroll
    for (int f = 0; f < 2; ++f)
      aq[f] = cat8(*(const v8bf*)(qr + f * 32 + hf * 8),
                   *(const v8bf*)(qr + f * 32 + 16 + hf * 8));
  }

  float m_i[8], l_i[8];
  v8f o[4] = {};
#pragma unroll
  for (int r = 0; r < 8; ++r) { m_i[r] = -3.0e38f; l_i[r] = 0.0f; }
  const float scale = 0.125f;                      // 1/sqrt(64)

  for (int kb = 0; kb < SEQ; kb += 32) {
    v8f sc[2];
#pragma unroll
    for (int tn = 0; tn < 2; ++tn) {
      const __bf16* kr = Kb + qkBase + (size_t)(kb + tn * 16 + lr) * DHEAD;
      v8f c = {};
#pragma unroll
      for (int f = 0; f < 2; ++f) {
        v16bf bk = cat8(*(const v8bf*)(kr + f * 32 + hf * 8),
                        *(const v8bf*)(kr + f * 32 + 16 + hf * 8));
        c = wmma_bf16(aq[f], bk, c);
      }
      sc[tn] = c;
    }
    float p0[8], p1[8], alpha[8];
#pragma unroll
    for (int r = 0; r < 8; ++r) {
      float s0 = sc[0][r] * scale;
      float s1 = sc[1][r] * scale;
      float mx = fmaxf(s0, s1);
#pragma unroll
      for (int off = 1; off < 16; off <<= 1)
        mx = fmaxf(mx, __shfl_xor(mx, off, 32));
      float m_new = fmaxf(m_i[r], mx);
      float e0 = __expf(s0 - m_new);
      float e1 = __expf(s1 - m_new);
      float rs = e0 + e1;
#pragma unroll
      for (int off = 1; off < 16; off <<= 1)
        rs += __shfl_xor(rs, off, 32);
      alpha[r] = __expf(m_i[r] - m_new);
      m_i[r]   = m_new;
      l_i[r]   = l_i[r] * alpha[r] + rs;
      p0[r] = e0; p1[r] = e1;
    }
#pragma unroll
    for (int dt = 0; dt < 4; ++dt)
#pragma unroll
      for (int r = 0; r < 8; ++r)
        o[dt][r] *= alpha[r];

    // P: C-layout -> A-layout via per-wave LDS tile (16x32)
    __bf16* pw = &Plds[wid][0];
#pragma unroll
    for (int r = 0; r < 8; ++r) {
      pw[(r + 8 * hf) * LDP + lr]      = f32_to_bf16(p0[r]);
      pw[(r + 8 * hf) * LDP + 16 + lr] = f32_to_bf16(p1[r]);
    }
    __syncthreads();
    v16bf ap;
    {
      const __bf16* pr = &Plds[wid][lr * LDP];
      ap = cat8(*(const v8bf*)(pr + hf * 8), *(const v8bf*)(pr + 16 + hf * 8));
    }
    __syncthreads();

#pragma unroll
    for (int dt = 0; dt < 4; ++dt) {
      const __bf16* vr = Vt + vtBase + (size_t)(dt * 16 + lr) * SEQ + kb + hf * 8;
      v16bf bvf = cat8(*(const v8bf*)vr, *(const v8bf*)(vr + 16));
      o[dt] = wmma_bf16(ap, bvf, o[dt]);
    }
  }

  const int b = bh >> 4, h = bh & 15;
#pragma unroll
  for (int dt = 0; dt < 4; ++dt) {
    const int d = dt * 16 + lr;
#pragma unroll
    for (int r = 0; r < 8; ++r) {
      const int s = q0 + r + 8 * hf;
      AO[((size_t)(b * SEQ + s)) * D_MODEL + h * DHEAD + d] =
          f32_to_bf16(o[dt][r] / l_i[r]);
    }
  }
}

// ---------------------------------------------------------------- launch
extern "C" void kernel_launch(void* const* d_in, const int* in_sizes, int n_in,
                              void* d_out, int out_size, void* d_ws, size_t ws_size,
                              hipStream_t stream) {
  (void)in_sizes; (void)n_in; (void)out_size; (void)ws_size;
  const float* x  = (const float*)d_in[0];
  const float* wq = (const float*)d_in[1];
  const float* bq = (const float*)d_in[2];
  const float* wk = (const float*)d_in[3];
  const float* bk = (const float*)d_in[4];
  const float* wv = (const float*)d_in[5];
  const float* bv = (const float*)d_in[6];
  const float* wo = (const float*)d_in[7];
  const float* bo = (const float*)d_in[8];
  float* out = (float*)d_out;

  char* ws = (char*)d_ws;
  const size_t M     = (size_t)BATCH * SEQ;          // 8192
  const size_t XB_SZ = M * D_MODEL * 2;              // 16 MiB
  __bf16* Xb   = (__bf16*)(ws);
  __bf16* Wqkv = (__bf16*)(ws + XB_SZ);
  __bf16* Wo   = (__bf16*)(ws + XB_SZ + 3u * D_MODEL * D_MODEL * 2);
  __bf16* Qb   = (__bf16*)(ws + XB_SZ + 4u * D_MODEL * D_MODEL * 2);
  __bf16* Kb   = (__bf16*)((char*)Qb + XB_SZ);
  __bf16* Vt   = (__bf16*)((char*)Kb + XB_SZ);
  __bf16* AO   = (__bf16*)((char*)Vt + XB_SZ);

  const int nx = (int)(M * D_MODEL);
  const int nw = D_MODEL * D_MODEL;
  cvt_f32_to_bf16<<<nx / 1024, 256, 0, stream>>>(x,  Xb, nx);
  cvt_f32_to_bf16<<<nw / 1024, 256, 0, stream>>>(wq, Wqkv,          nw);
  cvt_f32_to_bf16<<<nw / 1024, 256, 0, stream>>>(wk, Wqkv + nw,     nw);
  cvt_f32_to_bf16<<<nw / 1024, 256, 0, stream>>>(wv, Wqkv + 2 * nw, nw);
  cvt_f32_to_bf16<<<nw / 1024, 256, 0, stream>>>(wo, Wo, nw);

  // QKV: N=3072 -> 24 n-tiles, M=8192 -> 64 m-tiles
  gemm_qkv<<<dim3(24, 64), 256, 0, stream>>>(Xb, Wqkv, bq, bk, bv, Qb, Kb, Vt);

  // attention: 16 q-blocks of 128 rows, 64 (b,h) pairs
  attention<<<dim3(SEQ / 128, BATCH * HEADS), 256, 0, stream>>>(Qb, Kb, Vt, AO);

  // O projection: N=1024 -> 8 n-tiles
  gemm_out<<<dim3(8, 64), 256, 0, stream>>>(AO, Wo, bo, out);
}
